// I2S_78743930405363
// MI455X (gfx1250) — compile-verified, matrix operand order checked by reading, not run on hardware
//
#include <hip/hip_runtime.h>
#include <hip/hip_bf16.h>
#include <cstdint>

typedef __attribute__((ext_vector_type(16))) __bf16 v16bf;
typedef __attribute__((ext_vector_type(8)))  float  v8f;

__device__ __forceinline__ v8f wmma_bf16_16x16x32(v16bf a, v16bf b, v8f c) {
  // D = A(16x32 bf16) x B(32x16 bf16) + C(16x16 f32)
  return __builtin_amdgcn_wmma_f32_16x16x32_bf16(false, a, false, b, (short)0, c,
                                                 false, false);
}

// gfx1250 async DMA: copy 16 bytes global -> LDS per lane (ASYNCcnt-tracked).
// Inline asm per CDNA5 ISA / 15.18.3 opcode 98 (portable
// across ROCm-7.2 / amdgpu-toolchain, bypasses clang builtin arity skew).
__device__ __forceinline__ void async_g2lds_b128(uint32_t lds_byte_off,
                                                 const float* gptr) {
  asm volatile("global_load_async_to_lds_b128 %0, %1, off"
               :: "v"(lds_byte_off), "v"(gptr) : "memory");
}
__device__ __forceinline__ void wait_asynccnt0() {
  asm volatile("s_wait_asynccnt 0x0" ::: "memory");
}

#define BM 64
#define BN 32
#define BK 32
#define LDA_S 36   // padded LDS row stride (floats): 144B, 16B-aligned, odd/4 banks

// ---------------------------------------------------------------------------
// Tiled batched GEMM: C[z] = scale * (A[z] @ B[z]) (+bias[row]) (ReLU)
// A: M x K (row-major, lda), B: K x N (row-major, ldb), C: M x N (ldc), fp32.
// 256 threads = 8 waves; block computes BM x BN; wave w -> 16x16 tile
// (wm = w>>1, wn = w&1). K consumed BK=32 per step via bf16 WMMA, f32 acc.
// Interior+aligned tiles are filled with global_load_async_to_lds_b128;
// boundary/unaligned tiles use guarded scalar fills (zero padding).
// WMMA fragment layouts per CDNA5 ISA 7.12.2:
//   A 16x32 bf16: lane = (half<<4)|m ; element e -> K = (e&7)+((e>>3)<<4)+(half<<3)
//   B symmetric with lane holding column n = lane&15 ; C: elem r -> M=r+(half<<3)
// ---------------------------------------------------------------------------
__global__ void I2S_wmma_gemm(const float* __restrict__ A, const float* __restrict__ B,
                              float* __restrict__ C,
                              int M, int N, int K,
                              long lda, long ldb, long ldc,
                              long sA, long sB, long sC,
                              float scale, const float* __restrict__ bias, int relu)
{
  __shared__ float As[BM * LDA_S];
  __shared__ float Bs[BK * LDA_S];

  const int tid  = threadIdx.x;
  const int lane = tid & 31;
  const int wave = tid >> 5;
  const int wm   = wave >> 1;           // 0..3
  const int wn   = wave & 1;            // 0..1
  const int half = lane >> 4;
  const int lm   = lane & 15;

  const int m0 = blockIdx.y * BM;
  const int n0 = blockIdx.x * BN;

  const float* Ab = A + (long)blockIdx.z * sA;
  const float* Bb = B + (long)blockIdx.z * sB;
  float*       Cb = C + (long)blockIdx.z * sC;

  const int arow = (wm * 16 + lm) * LDA_S + half * 8;  // A fragment base (floats)
  const int bcol = wn * 16 + lm;                       // B fragment column

  v8f acc = {};
  for (int k0 = 0; k0 < K; k0 += BK) {
    // ---- stage A tile (BM x BK) ----
    const bool aAsync = ((lda & 3) == 0) && (m0 + BM <= M) && (k0 + BK <= K);
    if (aAsync) {
      for (int t = tid; t < (BM * BK) / 4; t += 256) {     // 512 b128 chunks
        const int r = t >> 3, ch = t & 7;
        const float* gp = Ab + (long)(m0 + r) * lda + k0 + ch * 4;
        async_g2lds_b128((uint32_t)(uintptr_t)&As[r * LDA_S + ch * 4], gp);
      }
    } else {
      for (int t = tid; t < BM * BK; t += 256) {
        const int r = t >> 5, c = t & 31;
        const int m = m0 + r, k = k0 + c;
        As[r * LDA_S + c] = (m < M && k < K) ? Ab[(long)m * lda + k] : 0.0f;
      }
    }
    // ---- stage B tile (BK x BN) ----
    const bool bAsync = ((ldb & 3) == 0) && (n0 + BN <= N) && (k0 + BK <= K);
    if (bAsync) {
      for (int t = tid; t < (BK * BN) / 4; t += 256) {     // 256 b128 chunks
        const int r = t >> 3, ch = t & 7;
        const float* gp = Bb + (long)(k0 + r) * ldb + n0 + ch * 4;
        async_g2lds_b128((uint32_t)(uintptr_t)&Bs[r * LDA_S + ch * 4], gp);
      }
    } else {
      for (int t = tid; t < BK * BN; t += 256) {
        const int r = t >> 5, c = t & 31;
        const int k = k0 + r, n = n0 + c;
        Bs[r * LDA_S + c] = (k < K && n < N) ? Bb[(long)k * ldb + n] : 0.0f;
      }
    }
    if (aAsync || bAsync) wait_asynccnt0();
    __syncthreads();

    // ---- fragments from LDS + WMMA ----
    const float4 a0 = *(const float4*)&As[arow];
    const float4 a1 = *(const float4*)&As[arow + 4];
    const float4 a2 = *(const float4*)&As[arow + 16];
    const float4 a3 = *(const float4*)&As[arow + 20];
    v16bf av, bv;
    av[0]  = (__bf16)a0.x; av[1]  = (__bf16)a0.y; av[2]  = (__bf16)a0.z; av[3]  = (__bf16)a0.w;
    av[4]  = (__bf16)a1.x; av[5]  = (__bf16)a1.y; av[6]  = (__bf16)a1.z; av[7]  = (__bf16)a1.w;
    av[8]  = (__bf16)a2.x; av[9]  = (__bf16)a2.y; av[10] = (__bf16)a2.z; av[11] = (__bf16)a2.w;
    av[12] = (__bf16)a3.x; av[13] = (__bf16)a3.y; av[14] = (__bf16)a3.z; av[15] = (__bf16)a3.w;
#pragma unroll
    for (int e = 0; e < 16; ++e) {
      const int kl = (e & 7) + ((e >> 3) << 4) + (half << 3);
      bv[e] = (__bf16)Bs[kl * LDA_S + bcol];
    }
    acc = wmma_bf16_16x16x32(av, bv, acc);
    __syncthreads();
  }

#pragma unroll
  for (int r = 0; r < 8; ++r) {
    const int m = m0 + wm * 16 + r + (half << 3);
    const int n = n0 + wn * 16 + lm;
    if (m < M && n < N) {
      float v = acc[r] * scale;
      if (bias) v += bias[m];
      if (relu) v = fmaxf(v, 0.0f);
      Cb[(long)m * ldc + n] = v;
    }
  }
}

// ---------------------------------------------------------------------------
// s2 -> so3 block-diagonal GEMM for one l (d = 2l+1):
//   out[b, g, off + u*d + m] = (1/sqrt(512)) * sum_f xs2[b, f, s+m] * psi[f, g, s+u]
// WMMA GEMM with rows = (b,m) [M=128*d], cols = (g,u) [N=8*d], K = 512.
// Operands are strided gathers (stride 49 / 392 floats) -> direct global loads.
// ---------------------------------------------------------------------------
__global__ void I2S_s2so3(const float* __restrict__ xs2, const float* __restrict__ psi,
                          float* __restrict__ xso3, int d, int s, int off, float scale)
{
  const int lane = threadIdx.x & 31;
  const int half = lane >> 4;
  const int lm   = lane & 15;
  const int Mrows = 128 * d, Ncols = 8 * d;
  const int m0 = blockIdx.y << 4;
  const int n0 = blockIdx.x << 4;

  const int  row = m0 + lm, col = n0 + lm;
  const bool rok = (row < Mrows), cok = (col < Ncols);
  const int  rowc = rok ? row : 0;
  const int  colc = cok ? col : 0;
  const int  b  = rowc / d, mm = rowc % d;
  const int  g  = colc / d, uu = colc % d;

  const float* Ap = xs2 + (long)b * (512 * 49) + (s + mm); // + k*49
  const float* Bp = psi + (long)g * 49 + (s + uu);         // + k*392

  v8f acc = {};
  for (int k0 = 0; k0 < 512; k0 += 32) {
    v16bf av, bv;
#pragma unroll
    for (int e = 0; e < 16; ++e) {
      const int k = k0 + (e & 7) + ((e >> 3) << 4) + (half << 3);
      float af = Ap[(long)k * 49];
      float bf = Bp[(long)k * 392];
      av[e] = (__bf16)(rok ? af : 0.0f);
      bv[e] = (__bf16)(cok ? bf : 0.0f);
    }
    acc = wmma_bf16_16x16x32(av, bv, acc);
  }

#pragma unroll
  for (int r = 0; r < 8; ++r) {
    const int rr = m0 + r + (half << 3);
    if (rr < Mrows && col < Ncols) {
      const int b2 = rr / d, m2 = rr % d;
      xso3[(long)b2 * 3640 + (long)g * 455 + off + uu * d + m2] = acc[r] * scale;
    }
  }
}

// ---------------------------------------------------------------------------
// so3 -> so3 conv (g output dim = 1, tiny contraction K = 8*d <= 104):
//   out[b, off + v*d + m] = (1/sqrt(8d)) * sum_{f,u} xa[b,f,off+u*d+m] * psi2[f,off+u*d+v]
// ---------------------------------------------------------------------------
__global__ void I2S_so3conv(const float* __restrict__ xa, const float* __restrict__ psi2,
                            float* __restrict__ out)
{
  const int idx = blockIdx.x * blockDim.x + threadIdx.x;
  if (idx >= 128 * 455) return;
  const int b = idx / 455;
  const int j = idx % 455;

  int off = 0, d = 1;
  for (int l = 0; l < 7; ++l) {
    d = 2 * l + 1;
    if (j < off + d * d) break;
    off += d * d;
  }
  const int r = j - off;
  const int v = r / d, m = r % d;

  const float* xb = xa + (long)b * 3640 + off;
  const float* pb = psi2 + off;
  float acc = 0.0f;
  for (int f = 0; f < 8; ++f)
    for (int u = 0; u < d; ++u)
      acc += xb[f * 455 + u * d + m] * pb[f * 455 + u * d + v];
  out[idx] = acc * rsqrtf(8.0f * (float)d);
}

// ---------------------------------------------------------------------------
extern "C" void kernel_launch(void* const* d_in, const int* in_sizes, int n_in,
                              void* d_out, int out_size, void* d_ws, size_t ws_size,
                              hipStream_t stream) {
  (void)in_sizes; (void)n_in; (void)out_size; (void)ws_size;

  const float* fmap     = (const float*)d_in[0];  // (128, 2048, 7, 7)
  const float* conv_w   = (const float*)d_in[1];  // (512, 2048)
  const float* conv_b   = (const float*)d_in[2];  // (512,)
  const float* proj_w   = (const float*)d_in[3];  // (7,7,192) -> (49,192)
  const float* proj_Y   = (const float*)d_in[4];  // (192, 49)
  const float* fs_w     = (const float*)d_in[5];  // (512, 8, 192) -> (4096,192)
  const float* fs_Y     = (const float*)d_in[6];  // (192, 49)
  const float* act_to   = (const float*)d_in[7];  // (455, 4000)
  const float* act_from = (const float*)d_in[8];  // (4000, 455)
  const float* so3_w    = (const float*)d_in[9];  // (8, 1, 192) -> (8,192)
  const float* so3_D    = (const float*)d_in[10]; // (192, 455)
  float* out = (float*)d_out;                     // (128, 1, 455)

  // workspace carve-up (floats), total ~46.6 MB
  float* ws   = (float*)d_ws;
  float* x1   = ws;                    // (128,512,49)   = 3211264
  float* xs2  = x1   + 3211264;        // (128,512,49)   = 3211264
  float* psi  = xs2  + 3211264;        // (512,8,49)     = 200704
  float* P    = psi  + 200704;         // (49,49)        = 2432 (padded)
  float* xso3 = P    + 2432;           // (128,8,455)    = 465920
  float* gbuf = xso3 + 465920;         // (128,8,4000)   = 4096000
  float* xa   = gbuf + 4096000;        // (128,8,455)    = 465920
  float* psi2 = xa   + 465920;         // (8,455)        = 3640

  const float RS192 = 0.07216878364870323f;   // 1/sqrt(192)
  const float RS512 = 0.04419417382415922f;   // 1/sqrt(512)
  const dim3 blk(256, 1, 1);
  const dim3 wblk(32, 1, 1);
  auto cdiv = [](int a, int b) { return (a + b - 1) / b; };

  // P = (proj_w @ proj_Y) / sqrt(192)                    [49 x 49, K=192]
  I2S_wmma_gemm<<<dim3(2, 1, 1), blk, 0, stream>>>(
      proj_w, proj_Y, P, 49, 49, 192, 192, 49, 49, 0, 0, 0, RS192, nullptr, 0);

  // psi = (fs_w @ fs_Y) / sqrt(192)                      [4096 x 49, K=192]
  I2S_wmma_gemm<<<dim3(2, 64, 1), blk, 0, stream>>>(
      fs_w, fs_Y, psi, 4096, 49, 192, 192, 49, 49, 0, 0, 0, RS192, nullptr, 0);

  // x1[b] = conv_w @ fmap[b] + conv_b                    [512 x 49, K=2048] x 128
  I2S_wmma_gemm<<<dim3(2, 8, 128), blk, 0, stream>>>(
      conv_w, fmap, x1, 512, 49, 2048, 2048, 49, 49,
      0, 2048L * 49, 512L * 49, 1.0f, conv_b, 0);

  // xs2 = x1 @ P   (projection hw->i, 1/sqrt(192) folded in P) [65536 x 49, K=49]
  I2S_wmma_gemm<<<dim3(2, 1024, 1), blk, 0, stream>>>(
      x1, P, xs2, 65536, 49, 49, 49, 49, 49, 0, 0, 0, 1.0f, nullptr, 0);

  // s2 -> so3 block-diagonal (per l)
  {
    int off = 0;
    for (int l = 0; l <= 6; ++l) {
      const int d = 2 * l + 1;
      I2S_s2so3<<<dim3(cdiv(8 * d, 16), cdiv(128 * d, 16), 1), wblk, 0, stream>>>(
          xs2, psi, xso3, d, l * l, off, RS512);
      off += d * d;
    }
  }

  // g = relu(xso3 @ act_to)                              [1024 x 4000, K=455]
  I2S_wmma_gemm<<<dim3(125, 16, 1), blk, 0, stream>>>(
      xso3, act_to, gbuf, 1024, 4000, 455, 455, 4000, 4000, 0, 0, 0, 1.0f, nullptr, 1);

  // xa = g @ act_from                                    [1024 x 455, K=4000]
  I2S_wmma_gemm<<<dim3(15, 16, 1), blk, 0, stream>>>(
      gbuf, act_from, xa, 1024, 455, 4000, 4000, 455, 455, 0, 0, 0, 1.0f, nullptr, 0);

  // psi2 = (so3_w @ so3_D) / sqrt(192)                   [8 x 455, K=192]
  I2S_wmma_gemm<<<dim3(15, 1, 1), blk, 0, stream>>>(
      so3_w, so3_D, psi2, 8, 455, 192, 192, 455, 455, 0, 0, 0, RS192, nullptr, 0);

  // so3 -> so3 conv, writes final output (128, 1, 455)
  I2S_so3conv<<<dim3(cdiv(128 * 455, 256), 1, 1), blk, 0, stream>>>(
      xa, psi2, out);
}
